// QValueNet_35699768164383
// MI455X (gfx1250) — matrix operand order repaired and checked by reading, not compiled, not run on hardware
//
#include <hip/hip_runtime.h>
#include <hip/hip_bf16.h>
#include <math.h>

typedef __attribute__((ext_vector_type(16))) _Float16 v16h;
typedef __attribute__((ext_vector_type(8)))  float    v8f;

#define FDIM 64

// ---------------------------------------------------------------------------
// Weight packing: f32 W[K,M] -> f16 in per-lane WMMA B layout.
// v16h entry index (kt*ntiles + c)*32 + lane holds, for element e:
//   W[(kt*32 + (lane>>4)*8 + (e<8 ? e : e+8)) * M + c*16 + (lane&15)]
// ---------------------------------------------------------------------------
__global__ void pack_w(const float* __restrict__ W, _Float16* __restrict__ Bp,
                       int K, int M) {
  long long t = (long long)blockIdx.x * blockDim.x + threadIdx.x;
  if (t >= (long long)K * M) return;
  const int e = (int)(t & 15);
  long long r = t >> 4;
  const int lane = (int)(r & 31); r >>= 5;
  const int ntiles = M >> 4;
  const int c  = (int)(r % ntiles);
  const int kt = (int)(r / ntiles);
  const int mcol = (c << 4) + (lane & 15);
  const int k = (kt << 5) + ((lane >> 4) << 3) + (e < 8 ? e : e + 8);
  Bp[t] = (_Float16)W[(long long)k * M + mcol];
}

// ---------------------------------------------------------------------------
// Fused GEMM: C[n, col] = act( (gather?)A[n,:K] @ W[K,M] + bias )
// One wave computes a 16x32 output strip (2 col tiles, shared A fetch).
// A per-lane data is two contiguous 8-float runs -> 4x b128 loads + cvt_pk.
// B comes pre-packed in f16 lane layout -> 2x b128 loads per tile.
// Row index is CLAMPED (not predicated) so loads stay branch-free; the
// bounds check lives only on the store.
// ---------------------------------------------------------------------------
template <int ACT>
__global__ void gemm_wmma(const float* __restrict__ A, int lda,
                          const long long* __restrict__ ridx,
                          const v16h* __restrict__ Bp,
                          const float* __restrict__ bias,
                          float* __restrict__ C, int ldc,
                          int nrows, int K, int M) {
  const int lane = threadIdx.x;
  const int m = lane & 15, g = lane >> 4;
  const int row0 = (blockIdx.x * blockDim.y + threadIdx.y) << 4;
  if (row0 >= nrows) return;  // wave-uniform

  int row = row0 + m;
  if (row >= nrows) row = nrows - 1;          // clamp: no divergent loads
  const long long ar = ridx ? ridx[row] : (long long)row;
  const float* ap = A + ar * (long long)lda + (g << 3);

  const int ntiles = M >> 4;
  const int c0 = blockIdx.y << 1;             // two col tiles per wave

  v8f acc0 = {0.f,0.f,0.f,0.f,0.f,0.f,0.f,0.f};
  v8f acc1 = {0.f,0.f,0.f,0.f,0.f,0.f,0.f,0.f};
  int kt = 0;
  for (int k0 = 0; k0 < K; k0 += 32, ++kt) {
    const float4 a0 = *reinterpret_cast<const float4*>(ap + k0);
    const float4 a1 = *reinterpret_cast<const float4*>(ap + k0 + 4);
    const float4 a2 = *reinterpret_cast<const float4*>(ap + k0 + 16);
    const float4 a3 = *reinterpret_cast<const float4*>(ap + k0 + 20);
    v16h av;
    av[0]  = (_Float16)a0.x; av[1]  = (_Float16)a0.y;
    av[2]  = (_Float16)a0.z; av[3]  = (_Float16)a0.w;
    av[4]  = (_Float16)a1.x; av[5]  = (_Float16)a1.y;
    av[6]  = (_Float16)a1.z; av[7]  = (_Float16)a1.w;
    av[8]  = (_Float16)a2.x; av[9]  = (_Float16)a2.y;
    av[10] = (_Float16)a2.z; av[11] = (_Float16)a2.w;
    av[12] = (_Float16)a3.x; av[13] = (_Float16)a3.y;
    av[14] = (_Float16)a3.z; av[15] = (_Float16)a3.w;

    const v16h bv0 = Bp[(size_t)(kt * ntiles + c0) * 32 + lane];
    const v16h bv1 = Bp[(size_t)(kt * ntiles + c0 + 1) * 32 + lane];
    acc0 = __builtin_amdgcn_wmma_f32_16x16x32_f16(false, av, false, bv0,
                                                  (short)0, acc0, false, false);
    acc1 = __builtin_amdgcn_wmma_f32_16x16x32_f16(false, av, false, bv1,
                                                  (short)0, acc1, false, false);
  }

#pragma unroll
  for (int tile = 0; tile < 2; ++tile) {
    const v8f acc = tile ? acc1 : acc0;
    const int ccol = ((c0 + tile) << 4) + m;
    const float bb = bias ? bias[ccol] : 0.f;
#pragma unroll
    for (int r = 0; r < 8; ++r) {
      const int rr = row0 + r + (g << 3);
      if (rr < nrows) {
        float val = acc[r] + bb;
        if (ACT == 1) val = val > 0.f ? val : 0.2f * val;
        C[(long long)rr * ldc + ccol] = val;
      }
    }
  }
}

__global__ void fill_zero(float* __restrict__ p, long long n) {
  long long i = (long long)blockIdx.x * blockDim.x + threadIdx.x;
  if (i < n) p[i] = 0.f;
}

// out[sidx[e]*64+f..f+3] += ew[e] * h[gidx[e]*64+f..f+3]
__global__ void edge_scatter(const float* __restrict__ h,
                             const long long* __restrict__ gidx,
                             const long long* __restrict__ sidx,
                             const float* __restrict__ ew,
                             float* __restrict__ out, long long E) {
  long long t = (long long)blockIdx.x * blockDim.x + threadIdx.x;
  if (t >= E * 16) return;
  const long long e = t >> 4;
  const int f = (int)(t & 15) << 2;
  const float wgt = ew[e];
  const float4 hv = *reinterpret_cast<const float4*>(h + gidx[e] * FDIM + f);
  float* op = out + sidx[e] * FDIM + f;
  atomicAdd(op + 0, wgt * hv.x);
  atomicAdd(op + 1, wgt * hv.y);
  atomicAdd(op + 2, wgt * hv.z);
  atomicAdd(op + 3, wgt * hv.w);
}

// out = lrelu(scat + selfh?) written with row stride ldo
__global__ void finish_conv(const float* __restrict__ scat,
                            const float* __restrict__ selfh,
                            float* __restrict__ out, int ldo, int n) {
  long long t = (long long)blockIdx.x * blockDim.x + threadIdx.x;
  if (t >= (long long)n * FDIM) return;
  const int node = (int)(t >> 6), f = (int)(t & 63);
  float v = scat[t] + (selfh ? selfh[t] : 0.f);
  v = v > 0.f ? v : 0.2f * v;
  out[(long long)node * ldo + f] = v;
}

// out[n,f] = in[n*ldi + f] + states[n]*a0
__global__ void preadd_states(const float* __restrict__ in, int ldi,
                              const float* __restrict__ states,
                              const float* __restrict__ a0,
                              float* __restrict__ out, int n) {
  long long t = (long long)blockIdx.x * blockDim.x + threadIdx.x;
  if (t >= (long long)n * FDIM) return;
  const int node = (int)(t >> 6), f = (int)(t & 63);
  out[t] = in[(long long)node * ldi + f] + states[node] * a0[0];
}

// Reference softmax is over a singleton axis -> all attention weights are 1.0;
// o[n, tq, :] = sum_tk v[n, tk, :] (identical for every tq).
__global__ void mha_sum_v(const float* __restrict__ v, float* __restrict__ o, int n) {
  long long t = (long long)blockIdx.x * blockDim.x + threadIdx.x;
  if (t >= (long long)n * FDIM) return;
  const int node = (int)(t >> 6), f = (int)(t & 63);
  const long long base = (long long)node * 192 + f;
  const float s = v[base] + v[base + 64] + v[base + 128];
  o[base] = s; o[base + 64] = s; o[base + 128] = s;
}

// s[n*3+t] = xa[n,t,:] . W + b
__global__ void pool_scores(const float* __restrict__ xa, const float* __restrict__ W,
                            const float* __restrict__ b, float* __restrict__ s, int n) {
  const int i = blockIdx.x * blockDim.x + threadIdx.x;
  if (i >= n * 3) return;
  const float* p = xa + (long long)i * 64;
  float acc = b[0];
#pragma unroll
  for (int d = 0; d < 64; ++d) acc += p[d] * W[d];
  s[i] = acc;
}

// out[n,f] = sum_t softmax(s[n,:])[t]*xa[n,t,f] + sum_t xs[n,t,f]
__global__ void pool_combine(const float* __restrict__ xa, const float* __restrict__ xs,
                             const float* __restrict__ s, float* __restrict__ out, int n) {
  long long t = (long long)blockIdx.x * blockDim.x + threadIdx.x;
  if (t >= (long long)n * FDIM) return;
  const int node = (int)(t >> 6), f = (int)(t & 63);
  const float s0 = s[node * 3], s1 = s[node * 3 + 1], s2 = s[node * 3 + 2];
  const float mx = fmaxf(s0, fmaxf(s1, s2));
  float w0 = __expf(s0 - mx), w1 = __expf(s1 - mx), w2 = __expf(s2 - mx);
  const float inv = 1.f / (w0 + w1 + w2);
  w0 *= inv; w1 *= inv; w2 *= inv;
  const long long base = (long long)node * 192 + f;
  out[t] = w0 * xa[base] + w1 * xa[base + 64] + w2 * xa[base + 128]
         + xs[base] + xs[base + 64] + xs[base + 128];
}

__global__ void seg_accum(const float* __restrict__ h, const long long* __restrict__ batch,
                          const float* __restrict__ states,
                          float* __restrict__ hsumg, float* __restrict__ hsg, int n) {
  long long t = (long long)blockIdx.x * blockDim.x + threadIdx.x;
  if (t >= (long long)n * FDIM) return;
  const int node = (int)(t >> 6), f = (int)(t & 63);
  const float val = h[t];
  const long long g = batch[node];
  atomicAdd(&hsumg[g * FDIM + f], val);
  if (states[node] == 1.0f) atomicAdd(&hsg[g * FDIM + f], val);
}

// out[n] = zz[n,:96] . W + b   (final fan_out==1 layer, plain VALU)
__global__ void final_dot(const float* __restrict__ zz, const float* __restrict__ W,
                          const float* __restrict__ b, float* __restrict__ out, int n) {
  const int i = blockIdx.x * blockDim.x + threadIdx.x;
  if (i >= n) return;
  const float* p = zz + (long long)i * 96;
  float acc = b[0];
#pragma unroll
  for (int k = 0; k < 96; ++k) acc += p[k] * W[k];
  out[i] = acc;
}

// ---------------------------------------------------------------------------
// Host orchestration
// ---------------------------------------------------------------------------
static _Float16* g_pack;  // set per-launch before use (stream-ordered reuse)

static inline dim3 g1d(long long n) { return dim3((unsigned)((n + 255) / 256)); }

static void gemm(hipStream_t st, const float* A, int lda, const long long* ridx,
                 const float* W, const float* b, float* C, int ldc,
                 int n, int K, int M, int act) {
  pack_w<<<g1d((long long)K * M), 256, 0, st>>>(W, g_pack, K, M);
  dim3 grid((n + 63) / 64, M / 32), block(32, 4);
  const v16h* Bp = reinterpret_cast<const v16h*>(g_pack);
  if (act) gemm_wmma<1><<<grid, block, 0, st>>>(A, lda, ridx, Bp, b, C, ldc, n, K, M);
  else     gemm_wmma<0><<<grid, block, 0, st>>>(A, lda, ridx, Bp, b, C, ldc, n, K, M);
}

extern "C" void kernel_launch(void* const* d_in, const int* in_sizes, int n_in,
                              void* d_out, int out_size, void* d_ws, size_t ws_size,
                              hipStream_t stream) {
  const int N = in_sizes[4];                 // states: (N,)
  const long long E = in_sizes[2];           // edge_weight: (E,)
  const int NT = N * 3;

  const float*     x      = (const float*)d_in[0];
  const long long* eidx   = (const long long*)d_in[1];
  const long long* esrc   = eidx;            // edge_index[0]
  const long long* edst   = eidx + E;        // edge_index[1]
  const float*     ew     = (const float*)d_in[2];
  const long long* batch  = (const long long*)d_in[3];
  const float*     states = (const float*)d_in[4];

  // params: jax pytree flatten order (dict keys sorted), base index 5
  auto P = [&](int i) { return (const float*)d_in[i]; };
  // 5..9  att1 {wk,wo,wq,ws,wv}; 10..14 att2; 15 beta0; 16 beta1; 17,18 beta2 W,b
  // 19..33 blocks1 [a0,a1W,a1b,convW,convb]x3; 34..48 blocks2
  // 49..54 enc_conv (W,b)x3; 55,56 enc_fc1; 57 enc_w1; 58 enc_w2
  // 59,60 fc1; 61,62 fc2; 63..65 gamma0..2; 66 gamma3; 67,68 gamma4

  // workspace arena (floats; every block a multiple of 64 floats -> 32B align)
  float* w = (float*)d_ws;
  size_t off = 0;
  auto alloc = [&](size_t nf) { float* p = w + off; off += nf; return p; };
  float* X1  = alloc((size_t)N * 64);
  float* X2  = alloc((size_t)N * 64);
  float* T1  = alloc((size_t)N * 64);
  float* S1  = alloc((size_t)N * 64);
  float* U   = alloc((size_t)N * 128);
  float* Y   = alloc((size_t)N * 128);
  float* A1s = alloc((size_t)N * 192);
  float* A2s = alloc((size_t)N * 192);
  float* V   = alloc((size_t)N * 192);
  float* O   = alloc((size_t)N * 192);
  float* XA  = alloc((size_t)N * 192);
  float* P3  = alloc((size_t)((N * 3 + 63) / 64) * 64);
  float* HGs = alloc((size_t)64 * 64);       // per-graph sums (G<=64)
  float* HGm = alloc((size_t)64 * 64);
  g_pack = (_Float16*)alloc((size_t)32768 / 2 * 2); // 32K floats = 64K halves
  float* HB = T1;                            // reuse after attention
  float* Z  = V;                             // reuse
  float* ZZ = O;                             // reuse

  const long long NE = (long long)N * 64;
  const long long E16 = E * 16;

  // ---- encoder: 3 shared-weight convs on both streams -------------------
  const float* cur1 = x + 64; int ld1 = 128;   // x1 = x[:, 64:]
  const float* cur2 = x;      int ld2 = 128;   // x2 = x[:, :64]
  for (int i = 0; i < 3; ++i) {
    const float* Wc = P(49 + 2 * i); const float* bc = P(50 + 2 * i);
    // x1: gather at dst, scatter to src
    gemm(stream, cur1, ld1, nullptr, Wc, bc, T1, 64, N, 64, 64, 0);
    fill_zero<<<g1d(NE), 256, 0, stream>>>(S1, NE);
    edge_scatter<<<g1d(E16), 256, 0, stream>>>(T1, edst, esrc, ew, S1, E);
    finish_conv<<<g1d(NE), 256, 0, stream>>>(S1, i > 0 ? T1 : nullptr, X1, 64, N);
    cur1 = X1; ld1 = 64;
    // x2: gather at src, scatter to dst
    gemm(stream, cur2, ld2, nullptr, Wc, bc, T1, 64, N, 64, 64, 0);
    fill_zero<<<g1d(NE), 256, 0, stream>>>(S1, NE);
    edge_scatter<<<g1d(E16), 256, 0, stream>>>(T1, esrc, edst, ew, S1, E);
    finish_conv<<<g1d(NE), 256, 0, stream>>>(S1, i > 0 ? T1 : nullptr, X2, 64, N);
    cur2 = X2; ld2 = 64;
  }
  // y = concat(x1@enc_w1, x2@enc_w2) @ enc_fc1 + b
  gemm(stream, X1, 64, nullptr, P(57), nullptr, U,      128, N, 64, 64, 0);
  gemm(stream, X2, 64, nullptr, P(58), nullptr, U + 64, 128, N, 64, 64, 0);
  gemm(stream, U, 128, nullptr, P(55), P(56),   Y,      128, N, 128, 128, 0);

  // ---- T residual blocks, stacked into [N, 3, 64] -----------------------
  auto run_blocks = [&](const float* h0, int ldh0, int pbase,
                        const long long* gi, const long long* si, float* xs) {
    const float* h = h0; int ldh = ldh0;
    for (int i = 0; i < 3; ++i) {
      const int b = pbase + i * 5;
      preadd_states<<<g1d(NE), 256, 0, stream>>>(h, ldh, states, P(b), T1, N);
      gemm(stream, T1, 64, nullptr, P(b + 1), P(b + 2), S1, 64, N, 64, 64, 1); // g
      gemm(stream, S1, 64, nullptr, P(b + 3), P(b + 4), T1, 64, N, 64, 64, 0); // conv-lin
      fill_zero<<<g1d(NE), 256, 0, stream>>>(S1, NE);
      edge_scatter<<<g1d(E16), 256, 0, stream>>>(T1, gi, si, ew, S1, E);
      finish_conv<<<g1d(NE), 256, 0, stream>>>(S1, nullptr, xs + i * 64, 192, N);
      h = xs + i * 64; ldh = 192;
    }
  };
  run_blocks(Y,      128, 19, edst, esrc, A1s);
  run_blocks(Y + 64, 128, 34, esrc, edst, A2s);

  // ---- attention (weights degenerate to 1.0) + softmax pooling ----------
  auto run_att = [&](const float* xs, int ab, int fcW, int fcb, float* outX) {
    gemm(stream, xs, 64, nullptr, P(ab + 4), nullptr, V, 64, NT, 64, 64, 0);   // v = xs@wv
    mha_sum_v<<<g1d(NE), 256, 0, stream>>>(V, O, N);
    gemm(stream, O, 64, nullptr, P(ab + 1), nullptr, XA, 64, NT, 64, 64, 0);   // @wo
    pool_scores<<<g1d(NT), 256, 0, stream>>>(XA, P(fcW), P(fcb), P3, N);
    pool_combine<<<g1d(NE), 256, 0, stream>>>(XA, xs, P3, outX, N);
  };
  run_att(A1s, 5,  59, 60, X1);
  run_att(A2s, 10, 61, 62, X2);

  // ---- beta combine ------------------------------------------------------
  gemm(stream, X1, 64, nullptr, P(15), nullptr, U,      128, N, 64, 64, 0);
  gemm(stream, X2, 64, nullptr, P(16), nullptr, U + 64, 128, N, 64, 64, 0);
  gemm(stream, U, 128, nullptr, P(17), P(18),   HB,     64,  N, 128, 64, 1);   // lrelu

  // ---- segment sums over batch ------------------------------------------
  fill_zero<<<g1d(64 * 64), 256, 0, stream>>>(HGs, 64 * 64);
  fill_zero<<<g1d(64 * 64), 256, 0, stream>>>(HGm, 64 * 64);
  seg_accum<<<g1d(NE), 256, 0, stream>>>(HB, batch, states, HGm, HGs, N);

  // ---- gamma head --------------------------------------------------------
  gemm(stream, HB,  64, nullptr, P(63), nullptr, Z,       192, N, 64, 64, 0);
  gemm(stream, HGs, 64, batch,   P(64), nullptr, Z + 64,  192, N, 64, 64, 0);  // hs[batch]
  gemm(stream, HGm, 64, batch,   P(65), nullptr, Z + 128, 192, N, 64, 64, 0);  // hsum[batch]
  gemm(stream, Z, 192, nullptr,  P(66), nullptr, ZZ,      96,  N, 192, 96, 1); // lrelu
  final_dot<<<g1d(N), 256, 0, stream>>>(ZZ, P(67), P(68), (float*)d_out, N);
}
